// GraphAttentionBlock_71554155152236
// MI455X (gfx1250) — matrix operand order repaired and checked by reading, not compiled, not run on hardware
//
#include <hip/hip_runtime.h>
#include <hip/hip_bf16.h>
#include <math.h>

typedef __attribute__((ext_vector_type(16))) _Float16 v16h;
typedef __attribute__((ext_vector_type(8)))  _Float16 v8h;
typedef __attribute__((ext_vector_type(8)))  float    v8f;

#define BB       2
#define NN       2048
#define HDIM     256
#define NHEADS   8
#define HEADDIM  32
#define FFDIM    1024
#define ROWS     (BB * NN)          // 4096
#define LN_EPS   1e-5f

// ---------------------------------------------------------------------------
// WMMA helpers (CDNA5 16x16x32 f16 -> f32, wave32).
// A-layout (16x32, M x K): lane<16 -> M=lane,   halfs = K{0..7,16..23}
//                          lane>=16 -> M=lane-16, halfs = K{8..15,24..31}
// B-layout (32x16, K x N): lane<16 -> N=lane,   halfs = K{0..7,16..23}
//                          lane>=16 -> N=lane-16, halfs = K{8..15,24..31}
// C-layout (16x16): VGPR r -> M = r + (lane>=16 ? 8:0), N = lane&15
// ---------------------------------------------------------------------------
__device__ __forceinline__ v16h load_a_tile(const _Float16* base, int ld, int lane) {
  const int m  = lane & 15;
  const int ko = (lane & 16) ? 8 : 0;
  const _Float16* p = base + (size_t)m * ld + ko;
  v8h lo = *(const v8h*)(p);        // K = ko .. ko+7
  v8h hi = *(const v8h*)(p + 16);   // K = ko+16 .. ko+23
  return __builtin_shufflevector(lo, hi, 0,1,2,3,4,5,6,7,8,9,10,11,12,13,14,15);
}

// Source must hold B's COLUMNS contiguously: row n of `base` (stride ld) is
// column n of B (i.e. K stored [key][dim], V stored transposed [dim][key],
// weights stored transposed [out][in]).
__device__ __forceinline__ v16h load_b_tile(const _Float16* base, int ld, int lane) {
  const int n  = lane & 15;
  const int ko = (lane & 16) ? 8 : 0;
  const _Float16* p = base + (size_t)n * ld + ko;
  v8h lo = *(const v8h*)(p);
  v8h hi = *(const v8h*)(p + 16);
  return __builtin_shufflevector(lo, hi, 0,1,2,3,4,5,6,7,8,9,10,11,12,13,14,15);
}

__device__ __forceinline__ v8f wmma_f16(v16h a, v16h b, v8f c) {
  return __builtin_amdgcn_wmma_f32_16x16x32_f16(false, a, false, b, (short)0, c,
                                                false, false);
}

__device__ __forceinline__ v8f vzero8() {
  v8f z = {0.f,0.f,0.f,0.f,0.f,0.f,0.f,0.f};
  return z;
}

// ---------------------------------------------------------------------------
// Weight transpose + f32 -> f16  (W[K x N] -> WT[N x K])
// ---------------------------------------------------------------------------
__global__ void transpose_to_f16(const float* __restrict__ src,
                                 _Float16* __restrict__ dst, int R, int C) {
  int i = blockIdx.x * blockDim.x + threadIdx.x;
  if (i >= R * C) return;
  int r = i / C, c = i - r * C;
  dst[(size_t)c * R + r] = (_Float16)src[i];
}

// ---------------------------------------------------------------------------
// LayerNorm over 256 cols, one wave per row, f16 output for WMMA consumption.
// ---------------------------------------------------------------------------
__global__ void __launch_bounds__(256) layernorm_f16(
    const float* __restrict__ x, const float* __restrict__ g,
    const float* __restrict__ bta, _Float16* __restrict__ y, int rows) {
  const int lane = threadIdx.x & 31;
  const int row  = blockIdx.x * (blockDim.x >> 5) + (threadIdx.x >> 5);
  if (row >= rows) return;
  const float* xr = x + (size_t)row * HDIM;
  float v[8], s = 0.f, s2 = 0.f;
#pragma unroll
  for (int j = 0; j < 8; ++j) {
    v[j] = xr[lane + j * 32];
    s += v[j]; s2 += v[j] * v[j];
  }
#pragma unroll
  for (int off = 16; off > 0; off >>= 1) {
    s  += __shfl_xor(s,  off, 32);
    s2 += __shfl_xor(s2, off, 32);
  }
  const float mu  = s  * (1.f / HDIM);
  const float var = s2 * (1.f / HDIM) - mu * mu;
  const float rs  = rsqrtf(var + LN_EPS);
  _Float16* yr = y + (size_t)row * HDIM;
#pragma unroll
  for (int j = 0; j < 8; ++j) {
    const int c = lane + j * 32;
    yr[c] = (_Float16)((v[j] - mu) * rs * g[c] + bta[c]);
  }
}

// ---------------------------------------------------------------------------
// Generic WMMA GEMM: out = A[M x K] * W + bias, W given transposed WT[N x K].
// One wave -> 16x64 output tile. Ping-pong double buffer, 2x unrolled: stage
// tiles are consumed by WMMAs then reloaded in place for step ks+2, so load
// latency hides under the MMA pipe with no register-rotation copies.
// ---------------------------------------------------------------------------
#define OP_HEADQK  0   // f16 out, reshaped per head [b,h,n,d]
#define OP_HEADVT  1   // f16 out, per head transposed [b,h,d,n]
#define OP_RES_F32 2   // f32 out = acc + bias + residual
#define OP_GELU_F16 3  // f16 out = gelu(acc + bias)

__global__ void __launch_bounds__(256) gemm_wmma(
    const _Float16* __restrict__ A, const _Float16* __restrict__ WT,
    const float* __restrict__ bias, const float* __restrict__ residual,
    float* __restrict__ outF, _Float16* __restrict__ outH,
    int M, int Nout, int K, int op) {
  const int lane = threadIdx.x & 31;
  const int wid  = blockIdx.x * (blockDim.x >> 5) + (threadIdx.x >> 5);
  const int ntiles = Nout >> 6;
  const int mt = wid / ntiles;
  const int nt = wid - mt * ntiles;
  if (mt * 16 >= M) return;

  v8f acc[4];
#pragma unroll
  for (int j = 0; j < 4; ++j) acc[j] = vzero8();

  const _Float16* Ab = A + (size_t)(mt * 16) * K;
  const _Float16* Wb = WT + (size_t)(nt * 64) * K;
  const int ksteps = K >> 5;   // 8 or 32 (even)

  auto lda = [&](int ks) { return load_a_tile(Ab + ks * 32, K, lane); };
  auto ldb = [&](int j, int ks) {
    return load_b_tile(Wb + (size_t)(j * 16) * K + ks * 32, K, lane);
  };

  // prologue: stages A (ks=0) and B (ks=1)
  v16h aA = lda(0), aB = lda(1);
  v16h bA[4], bB[4];
#pragma unroll
  for (int j = 0; j < 4; ++j) { bA[j] = ldb(j, 0); bB[j] = ldb(j, 1); }

  for (int ks = 0; ks < ksteps; ks += 2) {
    const int nA = (ks + 2 < ksteps) ? ks + 2 : ks;      // clamped tail (dead loads)
    const int nB = (ks + 3 < ksteps) ? ks + 3 : ks + 1;
#pragma unroll
    for (int j = 0; j < 4; ++j) acc[j] = wmma_f16(aA, bA[j], acc[j]);
    aA = lda(nA);
#pragma unroll
    for (int j = 0; j < 4; ++j) bA[j] = ldb(j, nA);
#pragma unroll
    for (int j = 0; j < 4; ++j) acc[j] = wmma_f16(aB, bB[j], acc[j]);
    aB = lda(nB);
#pragma unroll
    for (int j = 0; j < 4; ++j) bB[j] = ldb(j, nB);
  }

  const int nloc = lane & 15;
  const int mhi  = (lane & 16) ? 8 : 0;
#pragma unroll
  for (int j = 0; j < 4; ++j) {
    const int col = nt * 64 + j * 16 + nloc;
    const float bv = bias[col];
#pragma unroll
    for (int r = 0; r < 8; ++r) {
      const int row = mt * 16 + r + mhi;
      float v = acc[j][r] + bv;
      if (op == OP_RES_F32) {
        outF[(size_t)row * Nout + col] = v + residual[(size_t)row * Nout + col];
      } else if (op == OP_GELU_F16) {
        v = 0.5f * v * (1.f + erff(v * 0.70710678118654752f));
        outH[(size_t)row * Nout + col] = (_Float16)v;
      } else {
        const int h = col >> 5, d = col & 31;
        const int bb = row >> 11, n = row & (NN - 1);
        if (op == OP_HEADQK)
          outH[(((size_t)(bb * NHEADS + h)) * NN + n) * HEADDIM + d] = (_Float16)v;
        else  // OP_HEADVT
          outH[(((size_t)(bb * NHEADS + h)) * HEADDIM + d) * NN + n] = (_Float16)v;
      }
    }
  }
}

// ---------------------------------------------------------------------------
// Fused graph-attention, flash style. One wave = one (b,h) x 16 query rows.
// final_ij = e_ij * m_ij / sum_j(e_ij * m_ij),  e = exp(s + 2a-1 - rowmax),
// m = 0.25 + 0.75a   (the 1e-6 guard can never bind since sum >= 0.25).
// Adjacency is read exactly once from HBM; K-tiles ping-pong two iterations
// ahead, V/adjacency loads retire under the softmax VALU/trans work.
// ---------------------------------------------------------------------------
__global__ void __launch_bounds__(256) attn_wmma(
    const _Float16* __restrict__ q16, const _Float16* __restrict__ k16,
    const _Float16* __restrict__ vT16, const float* __restrict__ adj,
    _Float16* __restrict__ attn16) {
  __shared__ _Float16 plds[8 * 16 * 32];  // per-wave 16x32 P staging
  const int lane = threadIdx.x & 31;
  const int warp = threadIdx.x >> 5;
  const int wid  = blockIdx.x * 8 + warp;
  const int qt = wid & 127;       // N/16 query tiles
  const int bh = wid >> 7;        // b*8 + h
  const int h  = bh & 7;
  const int b  = bh >> 3;

  const v16h aq =
      load_a_tile(q16 + (((size_t)bh) * NN + qt * 16) * HEADDIM, HEADDIM, lane);

  const int nloc = lane & 15;
  const int mhi  = (lane & 16) ? 8 : 0;
  const float* adjrow = adj + ((size_t)b * NN + qt * 16 + mhi) * NN;

  v8f o0 = vzero8(), o1 = vzero8();
  float mrun[8], lrun[8];
#pragma unroll
  for (int r = 0; r < 8; ++r) { mrun[r] = -1e30f; lrun[r] = 0.f; }

  _Float16* slab = &plds[warp * (16 * 32)];
  const float scale = 0.17677669529663687f;  // 1/sqrt(HEADDIM)
  const _Float16* kbase = k16 + ((size_t)bh) * NN * HEADDIM;
  const _Float16* vbase = vT16 + ((size_t)bh) * HEADDIM * NN;
  const int nkt = NN / 32;  // 64 (even)

  auto ldk0 = [&](int kt) {
    return load_b_tile(kbase + (size_t)(kt * 32) * HEADDIM, HEADDIM, lane);
  };
  auto ldk1 = [&](int kt) {
    return load_b_tile(kbase + (size_t)(kt * 32 + 16) * HEADDIM, HEADDIM, lane);
  };

  // one key-tile (32 keys) of online-softmax attention
  auto body = [&](int kt, v16h bk0, v16h bk1) {
    const int k0 = kt * 32;
    // adjacency for this tile first: overlaps the S-MMAs
    float a0[8], a1[8];
#pragma unroll
    for (int r = 0; r < 8; ++r) {
      a0[r] = adjrow[(size_t)r * NN + k0 + nloc];
      a1[r] = adjrow[(size_t)r * NN + k0 + 16 + nloc];
    }

    v8f s0 = wmma_f16(aq, bk0, vzero8());
    v8f s1 = wmma_f16(aq, bk1, vzero8());

    // V tiles retire under the softmax work below
    const v16h bv0 = load_b_tile(vbase + k0, NN, lane);                   // dims 0..15
    const v16h bv1 = load_b_tile(vbase + (size_t)16 * NN + k0, NN, lane); // dims 16..31
    if (kt + 1 < nkt) {  // prefetch next adjacency tile (128B span per row)
#pragma unroll
      for (int r = 0; r < 8; ++r)
        __builtin_prefetch(&adjrow[(size_t)r * NN + k0 + 32 + 2 * nloc], 0, 0);
    }

    float p0[8], p1[8];
#pragma unroll
    for (int r = 0; r < 8; ++r) {
      s0[r] = s0[r] * scale + 2.f * a0[r] - 1.f;
      s1[r] = s1[r] * scale + 2.f * a1[r] - 1.f;
      p0[r] = 0.25f + 0.75f * a0[r];   // mask multiplier
      p1[r] = 0.25f + 0.75f * a1[r];
    }
#pragma unroll
    for (int r = 0; r < 8; ++r) {
      float tmax = fmaxf(s0[r], s1[r]);
#pragma unroll
      for (int off = 8; off > 0; off >>= 1)
        tmax = fmaxf(tmax, __shfl_xor(tmax, off, 32));   // stays in 16-lane half
      const float mnew = fmaxf(mrun[r], tmax);
      const float corr = __expf(mrun[r] - mnew);
      const float e0 = __expf(s0[r] - mnew) * p0[r];
      const float e1 = __expf(s1[r] - mnew) * p1[r];
      float ts = e0 + e1;
#pragma unroll
      for (int off = 8; off > 0; off >>= 1) ts += __shfl_xor(ts, off, 32);
      lrun[r] = lrun[r] * corr + ts;
      mrun[r] = mnew;
      o0[r] *= corr; o1[r] *= corr;
      p0[r] = e0; p1[r] = e1;
    }
    // C-layout P -> row-major LDS -> A-layout registers (wave-private slab;
    // per-wave LDS ops are in-order, explicit wait for safety).
#pragma unroll
    for (int r = 0; r < 8; ++r) {
      slab[(r + mhi) * 32 + nloc]      = (_Float16)p0[r];
      slab[(r + mhi) * 32 + 16 + nloc] = (_Float16)p1[r];
    }
    asm volatile("s_wait_dscnt 0" ::: "memory");
    const v16h ap = load_a_tile(slab, 32, lane);

    o0 = wmma_f16(ap, bv0, o0);
    o1 = wmma_f16(ap, bv1, o1);
  };

  // ping-pong K-tile stages two iterations ahead
  v16h kA0 = ldk0(0), kA1 = ldk1(0);
  v16h kB0 = ldk0(1), kB1 = ldk1(1);
  for (int kt = 0; kt < nkt; kt += 2) {
    body(kt, kA0, kA1);
    const int nA = (kt + 2 < nkt) ? kt + 2 : kt;
    kA0 = ldk0(nA); kA1 = ldk1(nA);
    body(kt + 1, kB0, kB1);
    const int nB = (kt + 3 < nkt) ? kt + 3 : kt + 1;
    kB0 = ldk0(nB); kB1 = ldk1(nB);
  }

#pragma unroll
  for (int r = 0; r < 8; ++r) {
    const float inv = 1.f / lrun[r];
    const int row = qt * 16 + r + mhi;
    _Float16* orow = attn16 + ((size_t)b * NN + row) * HDIM + h * HEADDIM;
    orow[nloc]      = (_Float16)(o0[r] * inv);
    orow[16 + nloc] = (_Float16)(o1[r] * inv);
  }
}

// ---------------------------------------------------------------------------
extern "C" void kernel_launch(void* const* d_in, const int* in_sizes, int n_in,
                              void* d_out, int out_size, void* d_ws, size_t ws_size,
                              hipStream_t stream) {
  const float* hidden = (const float*)d_in[0];
  const float* adjacency = (const float*)d_in[1];
  // d_in[2] positions: unused by the reference computation
  const float* wq = (const float*)d_in[3];   const float* bq = (const float*)d_in[4];
  const float* wk = (const float*)d_in[5];   const float* bk = (const float*)d_in[6];
  const float* wv = (const float*)d_in[7];   const float* bv = (const float*)d_in[8];
  const float* wo = (const float*)d_in[9];   const float* bo = (const float*)d_in[10];
  const float* g1 = (const float*)d_in[11];  const float* b1 = (const float*)d_in[12];
  const float* g2 = (const float*)d_in[13];  const float* b2 = (const float*)d_in[14];
  const float* wff1 = (const float*)d_in[15]; const float* bff1 = (const float*)d_in[16];
  const float* wff2 = (const float*)d_in[17]; const float* bff2 = (const float*)d_in[18];
  float* out = (float*)d_out;

  char* ws = (char*)d_ws;
  size_t off = 0;
  auto alloc = [&](size_t bytes) -> void* {
    void* p = ws + off;
    off = (off + bytes + 255) & ~(size_t)255;
    return p;
  };
  _Float16* xn16   = (_Float16*)alloc((size_t)ROWS * HDIM * 2);
  _Float16* q16    = (_Float16*)alloc((size_t)ROWS * HDIM * 2);
  _Float16* k16    = (_Float16*)alloc((size_t)ROWS * HDIM * 2);
  _Float16* vT16   = (_Float16*)alloc((size_t)ROWS * HDIM * 2);
  _Float16* attn16 = (_Float16*)alloc((size_t)ROWS * HDIM * 2);
  float*    hid2   = (float*)   alloc((size_t)ROWS * HDIM * 4);
  _Float16* h2_16  = (_Float16*)alloc((size_t)ROWS * HDIM * 2);
  _Float16* ff16   = (_Float16*)alloc((size_t)ROWS * FFDIM * 2);
  _Float16* wqT = (_Float16*)alloc((size_t)HDIM * HDIM * 2);
  _Float16* wkT = (_Float16*)alloc((size_t)HDIM * HDIM * 2);
  _Float16* wvT = (_Float16*)alloc((size_t)HDIM * HDIM * 2);
  _Float16* woT = (_Float16*)alloc((size_t)HDIM * HDIM * 2);
  _Float16* f1T = (_Float16*)alloc((size_t)HDIM * FFDIM * 2);
  _Float16* f2T = (_Float16*)alloc((size_t)FFDIM * HDIM * 2);

  // 1) weights -> transposed f16
  transpose_to_f16<<<(HDIM * HDIM + 255) / 256, 256, 0, stream>>>(wq, wqT, HDIM, HDIM);
  transpose_to_f16<<<(HDIM * HDIM + 255) / 256, 256, 0, stream>>>(wk, wkT, HDIM, HDIM);
  transpose_to_f16<<<(HDIM * HDIM + 255) / 256, 256, 0, stream>>>(wv, wvT, HDIM, HDIM);
  transpose_to_f16<<<(HDIM * HDIM + 255) / 256, 256, 0, stream>>>(wo, woT, HDIM, HDIM);
  transpose_to_f16<<<(HDIM * FFDIM + 255) / 256, 256, 0, stream>>>(wff1, f1T, HDIM, FFDIM);
  transpose_to_f16<<<(FFDIM * HDIM + 255) / 256, 256, 0, stream>>>(wff2, f2T, FFDIM, HDIM);

  // 2) LN1
  layernorm_f16<<<ROWS / 8, 256, 0, stream>>>(hidden, g1, b1, xn16, ROWS);

  // 3) Q/K/V projections (WMMA), K row-major per head, V transposed per head
  const int gBlk256 = (ROWS / 16) * (HDIM / 64) / 8;   // 128 blocks
  gemm_wmma<<<gBlk256, 256, 0, stream>>>(xn16, wqT, bq, nullptr, nullptr, q16,
                                         ROWS, HDIM, HDIM, OP_HEADQK);
  gemm_wmma<<<gBlk256, 256, 0, stream>>>(xn16, wkT, bk, nullptr, nullptr, k16,
                                         ROWS, HDIM, HDIM, OP_HEADQK);
  gemm_wmma<<<gBlk256, 256, 0, stream>>>(xn16, wvT, bv, nullptr, nullptr, vT16,
                                         ROWS, HDIM, HDIM, OP_HEADVT);

  // 4) fused masked attention (flash, single adjacency pass)
  attn_wmma<<<(BB * NHEADS * (NN / 16)) / 8, 256, 0, stream>>>(q16, k16, vT16,
                                                               adjacency, attn16);

  // 5) output projection + residual -> hidden2 (f32)
  gemm_wmma<<<gBlk256, 256, 0, stream>>>(attn16, woT, bo, hidden, hid2, nullptr,
                                         ROWS, HDIM, HDIM, OP_RES_F32);

  // 6) LN2
  layernorm_f16<<<ROWS / 8, 256, 0, stream>>>(hid2, g2, b2, h2_16, ROWS);

  // 7) FF1 + exact GELU -> f16
  const int gBlk1024 = (ROWS / 16) * (FFDIM / 64) / 8;  // 512 blocks
  gemm_wmma<<<gBlk1024, 256, 0, stream>>>(h2_16, f1T, bff1, nullptr, nullptr, ff16,
                                          ROWS, FFDIM, HDIM, OP_GELU_F16);

  // 8) FF2 + bias + residual -> final f32 output
  gemm_wmma<<<gBlk256, 256, 0, stream>>>(ff16, f2T, bff2, hid2, out, nullptr,
                                         ROWS, HDIM, FFDIM, OP_RES_F32);
}
